// BatteryMoEFlattenIntraCycleMoELayer_25357486916136
// MI455X (gfx1250) — matrix operand order repaired and checked by reading, not compiled, not run on hardware
//
#include <hip/hip_runtime.h>
#include <hip/hip_bf16.h>

// Problem constants (from reference)
#define B_    128
#define L_    100
#define NE    8
#define DM    512
#define K_RAW 900            // 3*CURVE_LEN
#define K_PAD 1024           // 32 WMMA k-steps = 8 TDM stages of 128
#define M_TOT (B_ * L_)      // 12800

#define STAGE_K   128                    // K elements per TDM stage
#define ROW_BYTES 272                    // 128*2 + 16 pad (bank-conflict-free, 16B aligned)
#define TILE_BYTES (64 * ROW_BYTES)      // 64 N-rows per stage

typedef __attribute__((ext_vector_type(16))) __bf16 bf16x16;
typedef __attribute__((ext_vector_type(8)))  float  f32x8;
typedef unsigned int u32x4 __attribute__((ext_vector_type(4)));
typedef int          i32x4 __attribute__((ext_vector_type(4)));
typedef int          i32x8 __attribute__((ext_vector_type(8)));

union Frag32 {             // 32 bytes = one WMMA bf16 A/B fragment per lane
    uint4   u[2];
    bf16x16 v;
};

// ---------------------------------------------------------------------------
// TDM: async-load one 64(N) x 128(K) bf16 tile of packed W into LDS.
// D# per CDNA5 ISA. 2D tensor: groups 2/3 unused (zero).
// LDS padding: pad_interval=64 dwords (one 256B row), pad_amount=4 dwords
// -> 272B row stride: ds_load_b128 lane n starts at bank 4n (conflict-free).
// ---------------------------------------------------------------------------
static __device__ __forceinline__ void tdm_load_b_tile(unsigned lds_off,
                                                       unsigned long long gaddr) {
    u32x4 g0;
    g0.x = 0x1u;                                        // count=1, user mode
    g0.y = lds_off;                                     // lds_addr (bytes)
    g0.z = (unsigned)(gaddr & 0xffffffffull);           // global_addr[31:0]
    g0.w = (unsigned)((gaddr >> 32) & 0x1ffffffull)     // global_addr[56:32]
         | (2u << 30);                                  // type=2 ("image")
    i32x8 g1;
    g1[0] = (1 << 16)                                   // data_size = 2 bytes
          | (1 << 20)                                   // pad_enable
          | (5 << 22)                                   // pad_interval: 64 dwords
          | (3 << 25);                                  // pad_amount: 4 dwords
    g1[1] = (int)((K_PAD & 0xffff) << 16);              // tensor_dim0[15:0] @63:48
    g1[2] = (int)(((K_PAD >> 16) & 0xffff)              // tensor_dim0[31:16]
          | ((DM & 0xffff) << 16));                     // tensor_dim1[15:0]
    g1[3] = (int)(((DM >> 16) & 0xffff)                 // tensor_dim1[31:16]
          | (STAGE_K << 16));                           // tile_dim0 = 128 (K)
    g1[4] = 64;                                         // tile_dim1 = 64 (N), tile_dim2=0
    g1[5] = K_PAD;                                      // tensor_dim0_stride[31:0]
    g1[6] = 0;                                          // stride hi / dim1_stride lo
    g1[7] = 0;
    i32x4 gz  = {0, 0, 0, 0};
#if defined(__clang_major__) && (__clang_major__ >= 23)
    i32x8 gz8 = {0, 0, 0, 0, 0, 0, 0, 0};
    __builtin_amdgcn_tensor_load_to_lds(g0, g1, gz, gz, gz8, 0);
#else
    __builtin_amdgcn_tensor_load_to_lds(g0, g1, gz, gz, 0);
#endif
}

// ---------------------------------------------------------------------------
// Pack X: f32 [M_TOT, 900] -> bf16 [M_TOT, 1024] (zero-padded K)
// ---------------------------------------------------------------------------
__global__ void pack_x_kernel(const float* __restrict__ x,
                              __hip_bfloat16* __restrict__ xp) {
    int idx = blockIdx.x * 256 + threadIdx.x;
    if (idx >= M_TOT * K_PAD) return;
    int row = idx >> 10;              // / K_PAD
    int k   = idx & (K_PAD - 1);
    float v = (k < K_RAW) ? x[(size_t)row * K_RAW + k] : 0.0f;
    xp[idx] = __float2bfloat16(v);
}

// ---------------------------------------------------------------------------
// Pack W: f32 [NE, 900, 512] -> bf16 [NE, 512, 1024]  (N-major, K contiguous)
// ---------------------------------------------------------------------------
__global__ void pack_w_kernel(const float* __restrict__ w,
                              __hip_bfloat16* __restrict__ wt) {
    int idx = blockIdx.x * 256 + threadIdx.x;
    if (idx >= NE * DM * K_PAD) return;
    int k = idx & (K_PAD - 1);
    int o = (idx >> 10) & (DM - 1);
    int e = idx >> 19;                // / (K_PAD*DM)
    float v = (k < K_RAW) ? w[((size_t)e * K_RAW + k) * DM + o] : 0.0f;
    wt[idx] = __float2bfloat16(v);
}

// ---------------------------------------------------------------------------
// Gates: masked renormalized softmax per batch row + guide loss reduction
// ---------------------------------------------------------------------------
__global__ void gates_kernel(const float* __restrict__ logits,
                             const int* __restrict__ masks,
                             float* __restrict__ gates,
                             __hip_bfloat16* __restrict__ guide_out) {
    __shared__ float red[B_];
    int b = threadIdx.x;
    float l[NE];
    float mx = -3.4e38f;
    #pragma unroll
    for (int e = 0; e < NE; ++e) { l[e] = logits[b * NE + e]; mx = fmaxf(mx, l[e]); }
    float s = 0.0f;
    #pragma unroll
    for (int e = 0; e < NE; ++e) { l[e] = __expf(l[e] - mx); s += l[e]; }
    float inv = 1.0f / s;
    float g[NE];
    float sum_masked = 0.0f, gsum = 0.0f;
    #pragma unroll
    for (int e = 0; e < NE; ++e) {
        float raw = l[e] * inv;
        float m = (masks[b * NE + e] == 1) ? 1.0f : 0.0f;
        g[e] = raw * m;
        sum_masked += raw * m;
        gsum += g[e];
    }
    float dn = 1.0f / (gsum + 1e-9f);
    #pragma unroll
    for (int e = 0; e < NE; ++e) gates[b * NE + e] = g[e] * dn;

    red[b] = sum_masked;
    for (int off = 64; off > 0; off >>= 1) {
        __syncthreads();
        if (b < off) red[b] += red[b + off];
    }
    __syncthreads();
    if (b == 0) {
        float t = 1.0f - red[0] / (float)B_;
        guide_out[0] = __float2bfloat16(t * t);
    }
}

// ---------------------------------------------------------------------------
// Fused MoE GEMM with TDM-staged weights.
// Block = 256 threads (8 wave32s); block tile = 128(M) x 64(N).
// Wave w owns rows R0 = bm*128 + w*16. K loop: 8 experts x 8 TDM stages x
// 4 WMMA k-steps. Wave 0 drives the TDM double-buffer pipeline; all waves
// consume B fragments from LDS (ds_load_b128, conflict-free via TDM pad).
// Per k-step: load all 4 B fragments into independent registers first, then
// fire 4 independent WMMAs -> single s_wait_dscnt per k-step, no XDL stalls.
// ---------------------------------------------------------------------------
__global__ __launch_bounds__(256)
void moe_gemm_kernel(const __hip_bfloat16* __restrict__ xp,   // [M_TOT, K_PAD]
                     const __hip_bfloat16* __restrict__ wt,   // [NE, DM, K_PAD]
                     const float* __restrict__ gates,         // [B_, NE]
                     const float* __restrict__ bias,          // [NE, DM]
                     __hip_bfloat16* __restrict__ out) {      // [M_TOT, DM]
    __shared__ __align__(16) unsigned char bTile[2][TILE_BYTES]; // 2 x 17KB
    __shared__ float gLds[B_ * NE];                              // 4 KB
    __shared__ float bLds[NE * 64];                              // 2 KB

    const int tid = threadIdx.x;
    const int bm  = blockIdx.x % (M_TOT / 128);
    const int bn  = blockIdx.x / (M_TOT / 128);
    const int N0  = bn * 64;

    for (int i = tid; i < B_ * NE; i += 256) gLds[i] = gates[i];
    for (int i = tid; i < NE * 64; i += 256) {
        int e = i >> 6, c = i & 63;
        bLds[i] = bias[e * DM + N0 + c];
    }

    const int wave = tid >> 5;
    const int lane = tid & 31;
    const int half = lane >> 4;
    const int mn   = lane & 15;          // M row (A) or N col (B/C)
    const int R0   = bm * 128 + wave * 16;

    // Batch index per accumulator element (row = R0 + i + 8*half; b = row/100)
    int brow[8];
    #pragma unroll
    for (int i = 0; i < 8; ++i) brow[i] = (R0 + i + 8 * half) / L_;

    f32x8 facc[4];
    #pragma unroll
    for (int t = 0; t < 4; ++t)
        #pragma unroll
        for (int i = 0; i < 8; ++i) facc[t][i] = 0.0f;

    // A fragment (16x32 bf16): elems 0..7 -> K0+8*half+0..7, 8..15 -> K0+16+8*half+0..7
    const __hip_bfloat16* aBase = xp + (size_t)(R0 + mn) * K_PAD + half * 8;

    const unsigned long long wtAddr = (unsigned long long)(uintptr_t)wt;
    const unsigned ldsOff[2] = { (unsigned)(uintptr_t)&bTile[0][0],
                                 (unsigned)(uintptr_t)&bTile[1][0] };

    // Prologue: wave 0 kicks off stage 0 (expert 0, k-chunk 0).
    if (wave == 0) {
        tdm_load_b_tile(ldsOff[0],
                        wtAddr + (unsigned long long)(N0 * K_PAD) * 2ull);
    }

    int q = 0;                           // flat stage index 0..63
    for (int e = 0; e < NE; ++e) {
        f32x8 acc[4];
        #pragma unroll
        for (int t = 0; t < 4; ++t)
            #pragma unroll
            for (int i = 0; i < 8; ++i) acc[t][i] = 0.0f;

        for (int st = 0; st < 8; ++st, ++q) {
            // Wait for stage q's tile, make it visible to all waves; this
            // barrier also guarantees everyone finished reading the buffer
            // that the next issue will overwrite.
            if (wave == 0) __builtin_amdgcn_s_wait_tensorcnt((short)0);
            __syncthreads();
            if (wave == 0 && q + 1 < NE * 8) {
                int eq = (q + 1) >> 3;
                int kq = ((q + 1) & 7) * STAGE_K;
                unsigned long long ga = wtAddr +
                    ((unsigned long long)((eq * DM + N0) * K_PAD + kq)) * 2ull;
                tdm_load_b_tile(ldsOff[(q + 1) & 1], ga);
            }

            const unsigned char* buf = &bTile[q & 1][0];
            const int kc = st * STAGE_K;

            #pragma unroll
            for (int ks = 0; ks < 4; ++ks) {
                Frag32 a;
                const __hip_bfloat16* ap = aBase + kc + ks * 32;
                a.u[0] = *(const uint4*)(ap);
                a.u[1] = *(const uint4*)(ap + 16);

                // Load all 4 B fragments first (independent regs), then the
                // 4 WMMAs issue back-to-back behind one dscnt wait.
                Frag32 bfr[4];
                #pragma unroll
                for (int t = 0; t < 4; ++t) {
                    // B frag (32x16): lane = N col, elems 0..7 -> k0+8*half+0..7, 8..15 -> k0+16+8*half+0..7
                    const unsigned char* bp = buf + (t * 16 + mn) * ROW_BYTES
                                                  + ks * 64 + half * 16;
                    bfr[t].u[0] = *(const uint4*)(bp);
                    bfr[t].u[1] = *(const uint4*)(bp + 32);
                }
                #pragma unroll
                for (int t = 0; t < 4; ++t) {
                    acc[t] = __builtin_amdgcn_wmma_f32_16x16x32_bf16(
                        false, a.v, false, bfr[t].v, (short)0, acc[t],
                        false, false);
                }
            }
        }

        // Fused per-expert epilogue: facc += gate[b(row), e] * (acc + bias[e,col])
        #pragma unroll
        for (int t = 0; t < 4; ++t) {
            float bv = bLds[e * 64 + t * 16 + mn];
            #pragma unroll
            for (int i = 0; i < 8; ++i) {
                float g = gLds[brow[i] * NE + e];
                facc[t][i] += g * (acc[t][i] + bv);
            }
        }
    }

    // Store bf16 output: element i -> row R0 + i + 8*half, col N0 + t*16 + mn
    #pragma unroll
    for (int t = 0; t < 4; ++t) {
        #pragma unroll
        for (int i = 0; i < 8; ++i) {
            int row = R0 + i + 8 * half;
            out[(size_t)row * DM + N0 + t * 16 + mn] = __float2bfloat16(facc[t][i]);
        }
    }
}

// ---------------------------------------------------------------------------
// Host launcher
// ---------------------------------------------------------------------------
extern "C" void kernel_launch(void* const* d_in, const int* in_sizes, int n_in,
                              void* d_out, int out_size, void* d_ws, size_t ws_size,
                              hipStream_t stream) {
    const float* x      = (const float*)d_in[0];  // [B, L, 3, 300] == [12800, 900]
    const float* logits = (const float*)d_in[1];  // [B, 8]
    const int*   masks  = (const int*)d_in[2];    // [B, 8]
    const float* W      = (const float*)d_in[3];  // [8, 900, 512]
    const float* bias   = (const float*)d_in[4];  // [8, 512]

    __hip_bfloat16* out = (__hip_bfloat16*)d_out; // [12800, 512] + guide loss tail

    // Workspace: gates f32 [0,4096) | Xp bf16 [12800,1024] | Wt bf16 [8,512,1024]
    char* ws = (char*)d_ws;
    float*          gates = (float*)ws;
    __hip_bfloat16* xp    = (__hip_bfloat16*)(ws + 4096);
    __hip_bfloat16* wtp   = (__hip_bfloat16*)(ws + 4096 + (size_t)M_TOT * K_PAD * 2);

    {
        int total = M_TOT * K_PAD;                      // 13,107,200
        pack_x_kernel<<<total / 256, 256, 0, stream>>>(x, xp);
    }
    {
        int total = NE * DM * K_PAD;                    // 4,194,304
        pack_w_kernel<<<total / 256, 256, 0, stream>>>(W, wtp);
    }
    gates_kernel<<<1, B_, 0, stream>>>(logits, masks, gates,
                                       out + (size_t)M_TOT * DM);
    {
        dim3 grid((M_TOT / 128) * (DM / 64));           // 800 blocks
        moe_gemm_kernel<<<grid, 256, 0, stream>>>(xp, wtp, gates, bias, out);
    }
}
